// Encoder_72249939853742
// MI455X (gfx1250) — matrix operand order repaired and checked by reading, not compiled, not run on hardware
//
#include <hip/hip_runtime.h>
#include <hip/hip_bf16.h>
#include <math.h>

// ---------------------------------------------------------------------------
// Problem constants (from reference): B=128, T=512, V=50000, NE=NF=NR=300,
// NC=64, gates G=4*NR=1200. Output depends only on batch row 127.
// ---------------------------------------------------------------------------
#define T_SEQ 512
#define NE    300
#define NF    300
#define NR    300
#define NG    1200          // 4*NR
#define NC    64
#define KP0   320           // K pad for 300 -> multiple of 32
#define NP0   304           // N pad for 300 -> multiple of 16
#define KPQ   608           // K pad for 600 -> multiple of 32
#define BROW  127           // batch row the output depends on

typedef __bf16 bf16;
typedef __attribute__((ext_vector_type(16))) __bf16 bf16x16;
typedef __attribute__((ext_vector_type(8)))  __bf16 bf16x8;
typedef __attribute__((ext_vector_type(8)))  float  floatx8;

__device__ __forceinline__ floatx8 wmma_bf16(bf16x16 a, bf16x16 b, floatx8 c) {
  // D = A(16x32 bf16) * B(32x16 bf16) + C(16x16 f32)
  return __builtin_amdgcn_wmma_f32_16x16x32_bf16(
      /*neg_a=*/false, a, /*neg_b=*/false, b,
      /*c_mod=*/(short)0, c, /*reuse_a=*/false, /*reuse_b=*/false);
}

// A fragment from padded bf16 row (row-major, Kpad stride).
// lane 0-15 : K = kc*32 + {0..7, 16..23}; lane 16-31: +8.
__device__ __forceinline__ bf16x16 load_a_frag(const bf16* __restrict__ row,
                                               int kc, int laneHi) {
  const bf16* p = row + kc * 32 + laneHi * 8;
  bf16x8 r1 = *(const bf16x8*)p;          // K +0..7
  bf16x8 r2 = *(const bf16x8*)(p + 16);   // K +16..23
  bf16x16 a;
#pragma unroll
  for (int i = 0; i < 8; ++i) { a[i] = r1[i]; a[i + 8] = r2[i]; }
  return a;
}

// B fragment from transposed bf16 weights WT[n][k] (Kpad stride).
// lane 0-15 : K = kc*32 + {0..7, 16..23}; lane 16-31: +8.
__device__ __forceinline__ bf16x16 load_b_frag(const bf16* __restrict__ col,
                                               int kc, int laneHi) {
  const bf16* p = col + kc * 32 + laneHi * 8;
  bf16x8 r1 = *(const bf16x8*)p;
  bf16x8 r2 = *(const bf16x8*)(p + 16);
  bf16x16 b;
#pragma unroll
  for (int i = 0; i < 8; ++i) { b[i] = r1[i]; b[i + 8] = r2[i]; }
  return b;
}

// 8 consecutive f32 -> bf16 halves of an A fragment, with K bounds (zero pad).
__device__ __forceinline__ void cvt8(const float* __restrict__ row, int k0,
                                     int K, bf16x16& a, int off) {
  if (k0 + 8 <= K) {
    float4 x = *(const float4*)(row + k0);
    float4 y = *(const float4*)(row + k0 + 4);
    a[off + 0] = (bf16)x.x; a[off + 1] = (bf16)x.y;
    a[off + 2] = (bf16)x.z; a[off + 3] = (bf16)x.w;
    a[off + 4] = (bf16)y.x; a[off + 5] = (bf16)y.y;
    a[off + 6] = (bf16)y.z; a[off + 7] = (bf16)y.w;
  } else {
#pragma unroll
    for (int i = 0; i < 8; ++i) {
      int k = k0 + i;
      a[off + i] = (k < K) ? (bf16)row[k] : (bf16)0.0f;
    }
  }
}

__device__ __forceinline__ float sigm(float x) { return 1.0f / (1.0f + expf(-x)); }

// ---------------------------------------------------------------------------
// Pack f32 weights [K][N] (row stride ldW) into transposed, padded bf16 [*][Kpad]
// ---------------------------------------------------------------------------
__global__ void pack_wT(const float* __restrict__ W, bf16* __restrict__ WT,
                        int K, int N, int Kpad, int ldW, int total) {
  int idx = blockIdx.x * blockDim.x + threadIdx.x;
  if (idx >= total) return;
  int n = idx / Kpad, k = idx % Kpad;
  float v = (n < N && k < K) ? W[(size_t)k * ldW + n] : 0.0f;
  WT[idx] = (bf16)v;
}

__global__ void zero16(bf16* __restrict__ p, int n) {
  int i = blockIdx.x * blockDim.x + threadIdx.x;
  if (i < n) p[i] = (bf16)0.0f;
}

// ---------------------------------------------------------------------------
// Kernel 1: hseq[t][k] = tanh(E[tokens[127,t]] @ W0 + b0)  (bf16, [512][320])
// Grid (8,19) x 128 threads: 4 waves/block, one 16x16 tile per wave.
// ---------------------------------------------------------------------------
__global__ __launch_bounds__(128)
void embed_gemm(const int* __restrict__ tokens, const float* __restrict__ E,
                const bf16* __restrict__ W0T, const float* __restrict__ b0,
                bf16* __restrict__ hseq) {
  int lane = threadIdx.x & 31, wid = threadIdx.x >> 5;
  int tile_m = blockIdx.x * 4 + wid;          // 0..31
  int tile_n = blockIdx.y;                    // 0..18
  int laneLo = lane & 15, laneHi = lane >> 4;
  int trow = tile_m * 16 + laneLo;            // time index 0..511
  int tok = tokens[BROW * T_SEQ + trow];
  const float* erow = E + (size_t)tok * NE;
  const bf16* bcol = W0T + (size_t)(tile_n * 16 + laneLo) * KP0;
  floatx8 acc = {};
  for (int kc = 0; kc < KP0 / 32; ++kc) {
    bf16x16 a;
    int kb = kc * 32 + laneHi * 8;
    cvt8(erow, kb, NE, a, 0);
    cvt8(erow, kb + 16, NE, a, 8);
    bf16x16 b = load_b_frag(bcol, kc, laneHi);
    acc = wmma_bf16(a, b, acc);
  }
  int col = tile_n * 16 + laneLo;
  if (col < NF) {
    float bias = b0[col];
#pragma unroll
    for (int v = 0; v < 8; ++v) {
      int row = tile_m * 16 + v + laneHi * 8;
      hseq[(size_t)row * KP0 + col] = (bf16)tanhf(acc[v] + bias);
    }
  }
}

// ---------------------------------------------------------------------------
// Kernel 2: time-parallel x-part of the gates for both directions:
//   gx[dir][t][j] = hseq[arow(t)] @ Wx + b    (f32, [2][512][1200])
// Backward direction reads the length-reversed input row.
// Grid (8,75,2) x 128 threads.
// ---------------------------------------------------------------------------
__global__ __launch_bounds__(128)
void gatesx_gemm(const bf16* __restrict__ hseq, const bf16* __restrict__ WfxT,
                 const bf16* __restrict__ WbxT, const float* __restrict__ bfv,
                 const float* __restrict__ bbv, const int* __restrict__ lengths,
                 float* __restrict__ gx) {
  int lane = threadIdx.x & 31, wid = threadIdx.x >> 5;
  int tile_m = blockIdx.x * 4 + wid;          // 0..31
  int tile_n = blockIdx.y;                    // 0..74
  int dir = blockIdx.z;                       // 0 fwd, 1 bwd
  int laneLo = lane & 15, laneHi = lane >> 4;
  int len = lengths[BROW];
  int t = tile_m * 16 + laneLo;
  int arow = (dir == 0) ? t : ((t < len) ? (len - 1 - t) : t);
  const bf16* Ar = hseq + (size_t)arow * KP0;
  const bf16* WT = dir ? WbxT : WfxT;
  const float* bias = dir ? bbv : bfv;
  int n = tile_n * 16 + laneLo;               // 0..1199
  const bf16* Bc = WT + (size_t)n * KP0;
  floatx8 acc = {};
  for (int kc = 0; kc < KP0 / 32; ++kc) {
    bf16x16 a = load_a_frag(Ar, kc, laneHi);
    bf16x16 b = load_b_frag(Bc, kc, laneHi);
    acc = wmma_bf16(a, b, acc);
  }
  float bv = bias[n];
  float* out = gx + (size_t)dir * T_SEQ * NG;
#pragma unroll
  for (int v = 0; v < 8; ++v) {
    int row = tile_m * 16 + v + laneHi * 8;
    out[(size_t)row * NG + n] = acc[v] + bv;
  }
}

// ---------------------------------------------------------------------------
// Kernel 3: sequential LSTM recurrence, batch=1, one block per direction.
// State (h,c) and gates live in LDS; per step each thread computes 1-2 gate
// columns of the 300x1200 matvec (coalesced f32 loads, L2-resident weights).
// Writes masked outputs directly into the padded q buffer (bwd re-reversed).
// ---------------------------------------------------------------------------
__global__ __launch_bounds__(1024)
void lstm_scan(const float* __restrict__ gx, const float* __restrict__ Wf,
               const float* __restrict__ Wb, const int* __restrict__ lengths,
               bf16* __restrict__ q) {
  int dir = blockIdx.x;
  const float* Wh = (dir ? Wb : Wf) + (size_t)NF * NG;  // recurrent rows 300..599
  const float* gin = gx + (size_t)dir * T_SEQ * NG;
  int len = lengths[BROW];
  __shared__ float h_s[NR];
  __shared__ float c_s[NR];
  __shared__ float g_s[NG];
  int tid = threadIdx.x;
  if (tid < NR) { h_s[tid] = 0.0f; c_s[tid] = 0.0f; }
  __syncthreads();
  for (int t = 0; t < T_SEQ; ++t) {
    for (int j = tid; j < NG; j += 1024) {
      float acc = gin[(size_t)t * NG + j];
      const float* wc = Wh + j;
#pragma unroll 4
      for (int k = 0; k < NR; ++k) acc = fmaf(h_s[k], wc[(size_t)k * NG], acc);
      g_s[j] = acc;
    }
    __syncthreads();
    if (tid < NR) {
      bool m = (t < len);
      float gi = g_s[tid], gg = g_s[NR + tid];
      float gf = g_s[2 * NR + tid], go = g_s[3 * NR + tid];
      float cn = sigm(gf + 1.0f) * c_s[tid] + sigm(gi) * fmaxf(gg, 0.0f);
      float hn = fmaxf(cn, 0.0f) * sigm(go);
      if (m) { c_s[tid] = cn; h_s[tid] = hn; }
      float y = m ? hn : 0.0f;
      int tq = dir ? (m ? (len - 1 - t) : t) : t;     // undo input reversal
      q[(size_t)tq * KPQ + dir * NR + tid] = (bf16)y;
    }
    __syncthreads();
  }
}

// ---------------------------------------------------------------------------
// Kernel 4: out[t][c] = q[t] @ P  ([512,608pad] x [608pad,64]).  Grid (8,4).
// ---------------------------------------------------------------------------
__global__ __launch_bounds__(128)
void out_gemm(const bf16* __restrict__ q, const bf16* __restrict__ PT,
              float* __restrict__ out) {
  int lane = threadIdx.x & 31, wid = threadIdx.x >> 5;
  int tile_m = blockIdx.x * 4 + wid;          // 0..31
  int tile_n = blockIdx.y;                    // 0..3
  int laneLo = lane & 15, laneHi = lane >> 4;
  const bf16* Ar = q + (size_t)(tile_m * 16 + laneLo) * KPQ;
  const bf16* Bc = PT + (size_t)(tile_n * 16 + laneLo) * KPQ;
  floatx8 acc = {};
  for (int kc = 0; kc < KPQ / 32; ++kc) {
    bf16x16 a = load_a_frag(Ar, kc, laneHi);
    bf16x16 b = load_b_frag(Bc, kc, laneHi);
    acc = wmma_bf16(a, b, acc);
  }
  int col = tile_n * 16 + laneLo;
#pragma unroll
  for (int v = 0; v < 8; ++v) {
    int row = tile_m * 16 + v + laneHi * 8;
    out[(size_t)row * NC + col] = acc[v];
  }
}

// ---------------------------------------------------------------------------
extern "C" void kernel_launch(void* const* d_in, const int* in_sizes, int n_in,
                              void* d_out, int out_size, void* d_ws, size_t ws_size,
                              hipStream_t stream) {
  const int*   tokens  = (const int*)d_in[0];
  const int*   lengths = (const int*)d_in[1];
  const float* E   = (const float*)d_in[2];
  const float* W0  = (const float*)d_in[3];
  const float* b0  = (const float*)d_in[4];
  const float* Wf  = (const float*)d_in[5];
  const float* bfv = (const float*)d_in[6];
  const float* Wb  = (const float*)d_in[7];
  const float* bbv = (const float*)d_in[8];
  const float* P   = (const float*)d_in[9];
  (void)in_sizes; (void)n_in; (void)out_size; (void)ws_size;

  char* ws = (char*)d_ws;
  auto take = [&](size_t bytes) -> char* {
    char* p = ws; ws += (bytes + 255) & ~(size_t)255; return p;
  };
  bf16*  W0T  = (bf16*)take((size_t)NP0 * KP0 * 2);   // [304][320]
  bf16*  WfxT = (bf16*)take((size_t)NG * KP0 * 2);    // [1200][320]
  bf16*  WbxT = (bf16*)take((size_t)NG * KP0 * 2);
  bf16*  PT   = (bf16*)take((size_t)NC * KPQ * 2);    // [64][608]
  bf16*  hseq = (bf16*)take((size_t)T_SEQ * KP0 * 2); // [512][320]
  bf16*  qbuf = (bf16*)take((size_t)T_SEQ * KPQ * 2); // [512][608]
  float* gx   = (float*)take((size_t)2 * T_SEQ * NG * 4);

  auto cdiv = [](int a, int b) { return (a + b - 1) / b; };

  // Pack transposed bf16 weights (x-parts only; recurrent rows stay f32).
  { int tot = NP0 * KP0; pack_wT<<<cdiv(tot, 256), 256, 0, stream>>>(W0, W0T, NE, NF, KP0, NF, tot); }
  { int tot = NG * KP0;  pack_wT<<<cdiv(tot, 256), 256, 0, stream>>>(Wf, WfxT, NF, NG, KP0, NG, tot); }
  { int tot = NG * KP0;  pack_wT<<<cdiv(tot, 256), 256, 0, stream>>>(Wb, WbxT, NF, NG, KP0, NG, tot); }
  { int tot = NC * KPQ;  pack_wT<<<cdiv(tot, 256), 256, 0, stream>>>(P,  PT, 2 * NR, NC, KPQ, NC, tot); }
  // Zero padded activation buffers (K-pad columns must read as 0).
  { int tot = T_SEQ * KP0; zero16<<<cdiv(tot, 256), 256, 0, stream>>>(hseq, tot); }
  { int tot = T_SEQ * KPQ; zero16<<<cdiv(tot, 256), 256, 0, stream>>>(qbuf, tot); }

  embed_gemm <<<dim3(8, 19),    128, 0, stream>>>(tokens, E, W0T, b0, hseq);
  gatesx_gemm<<<dim3(8, 75, 2), 128, 0, stream>>>(hseq, WfxT, WbxT, bfv, bbv, lengths, gx);
  lstm_scan  <<<2,              1024, 0, stream>>>(gx, Wf, Wb, lengths, qbuf);
  out_gemm   <<<dim3(8, 4),     128, 0, stream>>>(qbuf, PT, (float*)d_out);
}
